// NATLayer_63745904607566
// MI455X (gfx1250) — matrix operand order repaired and checked by reading, not compiled, not run on hardware
//
#include <hip/hip_runtime.h>
#include <hip/hip_bf16.h>
#include <math.h>

// ---------------------------------------------------------------------------
// Problem constants (match reference)
// ---------------------------------------------------------------------------
#define DIMC     1024
#define HEADS    16
#define HEAD_DIM 64
#define KS       7
#define HIDDEN   4096
#define BB       2
#define LL       4096
#define TOK      (BB * LL)          // 8192 rows
#define EPSF     1e-5f

typedef __bf16 bf16;
typedef __attribute__((ext_vector_type(16))) __bf16 v16bf;
typedef __attribute__((ext_vector_type(8)))  float  v8f;
typedef int v4i __attribute__((vector_size(16)));   // matches async builtin pointee

// CDNA5 async global->LDS path (gfx1250), guarded so either toolchain compiles
#if defined(__has_builtin)
#if __has_builtin(__builtin_amdgcn_global_load_async_to_lds_b128) && \
    __has_builtin(__builtin_amdgcn_s_wait_asynccnt)
#define HAVE_ASYNC_LDS 1
#endif
#endif
#ifndef HAVE_ASYNC_LDS
#define HAVE_ASYNC_LDS 0
#endif

// ---------------------------------------------------------------------------
// f32 -> bf16 cast (weights)
// ---------------------------------------------------------------------------
__global__ __launch_bounds__(256) void cast_f32_bf16(const float* __restrict__ in,
                                                     bf16* __restrict__ out, int n) {
    int i = blockIdx.x * 256 + threadIdx.x;
    if (i < n) out[i] = (bf16)in[i];
}

// ---------------------------------------------------------------------------
// LayerNorm (f32 stats) -> bf16 normalized output. One block per row.
// ---------------------------------------------------------------------------
__global__ __launch_bounds__(256) void layernorm_bf16(const float* __restrict__ x,
                                                      const float* __restrict__ w,
                                                      const float* __restrict__ b,
                                                      bf16* __restrict__ out) {
    __shared__ float rs[256], rs2[256];
    const int row = blockIdx.x;
    const float* xr = x + (size_t)row * DIMC;
    float s = 0.f, s2 = 0.f;
    #pragma unroll
    for (int i = threadIdx.x; i < DIMC; i += 256) {
        float v = xr[i];
        s += v; s2 += v * v;
    }
    rs[threadIdx.x] = s; rs2[threadIdx.x] = s2;
    __syncthreads();
    #pragma unroll
    for (int st = 128; st > 0; st >>= 1) {
        if (threadIdx.x < st) {
            rs[threadIdx.x]  += rs[threadIdx.x + st];
            rs2[threadIdx.x] += rs2[threadIdx.x + st];
        }
        __syncthreads();
    }
    const float mean = rs[0] * (1.0f / DIMC);
    const float var  = rs2[0] * (1.0f / DIMC) - mean * mean;
    const float rstd = rsqrtf(var + EPSF);
    bf16* orow = out + (size_t)row * DIMC;
    #pragma unroll
    for (int i = threadIdx.x; i < DIMC; i += 256) {
        orow[i] = (bf16)((xr[i] - mean) * rstd * w[i] + b[i]);
    }
}

// ---------------------------------------------------------------------------
// bf16 WMMA GEMM: C = A(MxK) * B(KxN) + bias [+resid] [GELU], f32 accumulate.
// Block tile 128x128, KC=32, 256 threads = 8 waves (4 M x 2 N), 32x64/wave
// = 2x4 v_wmma_f32_16x16x32_bf16 tiles. Double-buffered LDS; A tile staged
// with GLOBAL_LOAD_ASYNC_TO_LDS_B128 when available. All dims tile-exact.
// ---------------------------------------------------------------------------
#define BM 128
#define BN 128
#define KC 32
#define LDA 40   // bf16: 80B row stride (16B aligned for B128 async writes)
#define LDB 34   // bf16: odd-dword stride for transposed B tile

template <bool GELU, bool RESID, bool OUTF32>
__global__ __launch_bounds__(256) void gemm_bf16_wmma(
        const bf16* __restrict__ A, const bf16* __restrict__ Bm,
        const float* __restrict__ bias, const float* __restrict__ resid,
        void* __restrict__ outp, int M, int N, int K) {
    __shared__ bf16 As[2][BM][LDA];   // [m][k], k contiguous, XOR-swizzled 16B segs
    __shared__ bf16 Bs[2][BN][LDB];   // [n][k], k contiguous (transposed on store)
    (void)M;

    const int tid   = threadIdx.x;
    const int lane  = tid & 31;
    const int wid   = tid >> 5;           // 0..7
    const int waveM = wid >> 1;           // 0..3 -> 32 rows each
    const int waveN = wid & 1;            // 0..1 -> 64 cols each
    const int hh    = lane >> 4;          // half-lane select
    const int lm    = lane & 15;

    const int blockCol = blockIdx.x * BN;
    const int blockRow = blockIdx.y * BM;

    v8f acc[2][4];
    const v8f vz = {0.f,0.f,0.f,0.f,0.f,0.f,0.f,0.f};
    #pragma unroll
    for (int tm = 0; tm < 2; ++tm)
        #pragma unroll
        for (int tn = 0; tn < 4; ++tn) acc[tm][tn] = vz;

    // ---- A staging: 128x32 tile = 512 x 16B chunks, 2 per thread ----
    auto stageA = [&](int k0, int b) {
        #pragma unroll
        for (int i = 0; i < 2; ++i) {
            const int c   = tid + 256 * i;
            const int row = c >> 2;
            const int seg = c & 3;
            const int ss  = seg ^ (row & 3);          // bank swizzle
            const bf16* g = A + (size_t)(blockRow + row) * K + k0 + seg * 8;
            bf16* l = &As[b][row][ss * 8];
#if HAVE_ASYNC_LDS
            __builtin_amdgcn_global_load_async_to_lds_b128(
                (__attribute__((address_space(1))) v4i*)g,
                (__attribute__((address_space(3))) v4i*)l, 0, 0);
#else
            #pragma unroll
            for (int e = 0; e < 8; ++e) l[e] = g[e];
#endif
        }
    };
    auto waitA = [&]() {
#if HAVE_ASYNC_LDS
        __builtin_amdgcn_s_wait_asynccnt(0);
#endif
    };

    // ---- B staging: 32x128 tile, transposed into [n][k] ----
    const int bkb = tid >> 5;             // 0..7
    const int bnb = (tid & 31) * 4;       // 4 bf16 per k step
    auto loadB = [&](int k0, bf16* breg) {
        #pragma unroll
        for (int kk = 0; kk < 4; ++kk) {
            const bf16* src = Bm + (size_t)(k0 + bkb + kk * 8) * N + blockCol + bnb;
            #pragma unroll
            for (int e = 0; e < 4; ++e) breg[kk * 4 + e] = src[e];
        }
    };
    auto storeB = [&](const bf16* breg, int b) {
        #pragma unroll
        for (int kk = 0; kk < 4; ++kk) {
            const int k = bkb + kk * 8;
            #pragma unroll
            for (int e = 0; e < 4; ++e) Bs[b][bnb + e][k] = breg[kk * 4 + e];
        }
    };

    // ---- fragment loads (documented CDNA5 VGPR layouts) + 8 WMMAs ----
    auto compute = [&](int b) {
        v16bf af[2], bfr[4];
        #pragma unroll
        for (int tm = 0; tm < 2; ++tm) {
            const int mrow = waveM * 32 + tm * 16 + lm;
            const int sw   = mrow & 3;
            #pragma unroll
            for (int j = 0; j < 8; ++j) {
                const int k   = ((j >> 2) << 4) + (hh << 3) + ((j & 3) << 1);
                const int idx = (((k >> 3) ^ sw) << 3) + (k & 7);
                af[tm][2 * j]     = As[b][mrow][idx];
                af[tm][2 * j + 1] = As[b][mrow][idx + 1];
            }
        }
        #pragma unroll
        for (int tn = 0; tn < 4; ++tn) {
            const int ncol = waveN * 64 + tn * 16 + lm;
            #pragma unroll
            for (int j = 0; j < 8; ++j) {
                const int k = ((j >> 2) << 4) + (hh << 3) + ((j & 3) << 1);
                bfr[tn][2 * j]     = Bs[b][ncol][k];
                bfr[tn][2 * j + 1] = Bs[b][ncol][k + 1];
            }
        }
        #pragma unroll
        for (int tm = 0; tm < 2; ++tm)
            #pragma unroll
            for (int tn = 0; tn < 4; ++tn)
                acc[tm][tn] = __builtin_amdgcn_wmma_f32_16x16x32_bf16(
                    false, af[tm], false, bfr[tn], (short)0, acc[tm][tn],
                    false, false);
    };

    // ---- software pipeline: double-buffered LDS ----
    {
        bf16 breg[16];
        loadB(0, breg);
        stageA(0, 0);
        storeB(breg, 0);
        waitA();
    }
    __syncthreads();

    int buf = 0;
    for (int k0 = 0; k0 < K; k0 += KC) {
        bf16 breg[16];
        const bool nxt = (k0 + KC) < K;
        if (nxt) {
            loadB(k0 + KC, breg);        // global loads overlap compute
            stageA(k0 + KC, buf ^ 1);    // async DMA overlaps compute
        }
        compute(buf);
        if (nxt) {
            storeB(breg, buf ^ 1);
            waitA();
        }
        __syncthreads();
        buf ^= 1;
    }

    // ---- epilogue ----
    #pragma unroll
    for (int tm = 0; tm < 2; ++tm) {
        #pragma unroll
        for (int tn = 0; tn < 4; ++tn) {
            const int mbase = blockRow + waveM * 32 + tm * 16;
            const int n     = blockCol + waveN * 64 + tn * 16 + lm;
            const float bn  = bias[n];
            #pragma unroll
            for (int r = 0; r < 8; ++r) {
                const int m = mbase + r + hh * 8;   // C layout: lanes16-31 hold M=r+8
                float v = acc[tm][tn][r] + bn;
                if (RESID) v += resid[(size_t)m * N + n];
                if (GELU)  v = 0.5f * v * (1.0f + erff(v * 0.70710678118654752f));
                if (OUTF32) ((float*)outp)[(size_t)m * N + n] = v;
                else        ((bf16*)outp)[(size_t)m * N + n] = (bf16)v;
            }
        }
    }
}

// ---------------------------------------------------------------------------
// Neighborhood attention (KS=7) with relative position bias, exact softmax.
// One thread per (b, l, head); qkv is bf16 [TOK][3*DIM] laid out as
// [3][HEADS][HEAD_DIM] per row. Compute-light: ~0.24 GFLOP total.
// ---------------------------------------------------------------------------
__global__ __launch_bounds__(256) void nat_attention(const bf16* __restrict__ qkv,
                                                     const float* __restrict__ rpb,
                                                     bf16* __restrict__ out) {
    const int idx = blockIdx.x * 256 + threadIdx.x;
    if (idx >= TOK * HEADS) return;
    const int h = idx & (HEADS - 1);
    const int t = idx >> 4;            // b*L + l
    const int l = t & (LL - 1);
    const int tb = t - l;              // b*L

    int start = l - (KS / 2);
    if (start < 0) start = 0;
    if (start > LL - KS) start = LL - KS;

    const size_t rstride = 3 * DIMC;
    const bf16* qp = qkv + (size_t)t * rstride + h * HEAD_DIM;

    float q[HEAD_DIM];
    #pragma unroll
    for (int d = 0; d < HEAD_DIM; ++d) q[d] = (float)qp[d] * 0.125f; // 64^-0.5

    float sc[KS];
    #pragma unroll
    for (int j = 0; j < KS; ++j) {
        const bf16* kp = qkv + (size_t)(tb + start + j) * rstride + DIMC + h * HEAD_DIM;
        float s = 0.f;
        #pragma unroll
        for (int d = 0; d < HEAD_DIM; ++d) s += q[d] * (float)kp[d];
        const int rel = start + j - l + (KS - 1);
        sc[j] = s + rpb[h * (2 * KS - 1) + rel];
    }
    float mx = sc[0];
    #pragma unroll
    for (int j = 1; j < KS; ++j) mx = fmaxf(mx, sc[j]);
    float den = 0.f;
    #pragma unroll
    for (int j = 0; j < KS; ++j) { sc[j] = expf(sc[j] - mx); den += sc[j]; }
    const float inv = 1.0f / den;

    float o[HEAD_DIM];
    #pragma unroll
    for (int d = 0; d < HEAD_DIM; ++d) o[d] = 0.f;
    #pragma unroll
    for (int j = 0; j < KS; ++j) {
        const bf16* vp = qkv + (size_t)(tb + start + j) * rstride + 2 * DIMC + h * HEAD_DIM;
        const float wj = sc[j] * inv;
        #pragma unroll
        for (int d = 0; d < HEAD_DIM; ++d) o[d] += wj * (float)vp[d];
    }
    bf16* op = out + (size_t)t * DIMC + h * HEAD_DIM;
    #pragma unroll
    for (int d = 0; d < HEAD_DIM; ++d) op[d] = (bf16)o[d];
}

// ---------------------------------------------------------------------------
// Launcher
// ---------------------------------------------------------------------------
extern "C" void kernel_launch(void* const* d_in, const int* in_sizes, int n_in,
                              void* d_out, int out_size, void* d_ws, size_t ws_size,
                              hipStream_t stream) {
    const float* x      = (const float*)d_in[0];
    const float* n1w    = (const float*)d_in[1];
    const float* n1b    = (const float*)d_in[2];
    const float* qkvw   = (const float*)d_in[3];
    const float* qkvb   = (const float*)d_in[4];
    const float* rpb    = (const float*)d_in[5];
    const float* projw  = (const float*)d_in[6];
    const float* projb  = (const float*)d_in[7];
    const float* n2w    = (const float*)d_in[8];
    const float* n2b    = (const float*)d_in[9];
    const float* fc1w   = (const float*)d_in[10];
    const float* fc1b   = (const float*)d_in[11];
    const float* fc2w   = (const float*)d_in[12];
    const float* fc2b   = (const float*)d_in[13];
    float* out = (float*)d_out;

    // workspace carve-up (256B aligned)
    char* base = (char*)d_ws;
    size_t off = 0;
    auto carve = [&](size_t bytes) -> char* {
        char* p = base + off;
        off = (off + bytes + 255) & ~(size_t)255;
        return p;
    };
    bf16*  qkvw_b = (bf16*)carve((size_t)DIMC * 3 * DIMC * sizeof(bf16));
    bf16*  projw_b= (bf16*)carve((size_t)DIMC * DIMC * sizeof(bf16));
    bf16*  fc1w_b = (bf16*)carve((size_t)DIMC * HIDDEN * sizeof(bf16));
    bf16*  fc2w_b = (bf16*)carve((size_t)HIDDEN * DIMC * sizeof(bf16));
    bf16*  xn     = (bf16*)carve((size_t)TOK * DIMC * sizeof(bf16));
    bf16*  qkv    = (bf16*)carve((size_t)TOK * 3 * DIMC * sizeof(bf16));
    bf16*  attn   = (bf16*)carve((size_t)TOK * DIMC * sizeof(bf16));
    float* x2     = (float*)carve((size_t)TOK * DIMC * sizeof(float));
    bf16*  mb     = (bf16*)carve((size_t)TOK * DIMC * sizeof(bf16));
    bf16*  hb     = (bf16*)carve((size_t)TOK * HIDDEN * sizeof(bf16));
    (void)ws_size; (void)n_in; (void)in_sizes; (void)out_size;

    // 1) cast weights to bf16
    {
        int n;
        n = DIMC * 3 * DIMC; cast_f32_bf16<<<(n + 255) / 256, 256, 0, stream>>>(qkvw, qkvw_b, n);
        n = DIMC * DIMC;     cast_f32_bf16<<<(n + 255) / 256, 256, 0, stream>>>(projw, projw_b, n);
        n = DIMC * HIDDEN;   cast_f32_bf16<<<(n + 255) / 256, 256, 0, stream>>>(fc1w, fc1w_b, n);
        n = HIDDEN * DIMC;   cast_f32_bf16<<<(n + 255) / 256, 256, 0, stream>>>(fc2w, fc2w_b, n);
    }

    // 2) LN1 -> bf16
    layernorm_bf16<<<TOK, 256, 0, stream>>>(x, n1w, n1b, xn);

    // 3) qkv = xn @ qkv_w + qkv_b          (8192 x 3072, K=1024) -> bf16
    gemm_bf16_wmma<false, false, false>
        <<<dim3((3 * DIMC) / BN, TOK / BM), 256, 0, stream>>>(
            xn, qkvw_b, qkvb, nullptr, (void*)qkv, TOK, 3 * DIMC, DIMC);

    // 4) neighborhood attention -> bf16
    nat_attention<<<(TOK * HEADS) / 256, 256, 0, stream>>>(qkv, rpb, attn);

    // 5) x2 = x + attn @ proj_w + proj_b   (8192 x 1024, K=1024) -> f32
    gemm_bf16_wmma<false, true, true>
        <<<dim3(DIMC / BN, TOK / BM), 256, 0, stream>>>(
            attn, projw_b, projb, x, (void*)x2, TOK, DIMC, DIMC);

    // 6) LN2 -> bf16
    layernorm_bf16<<<TOK, 256, 0, stream>>>(x2, n2w, n2b, mb);

    // 7) hb = gelu(mb @ fc1_w + fc1_b)     (8192 x 4096, K=1024) -> bf16
    gemm_bf16_wmma<true, false, false>
        <<<dim3(HIDDEN / BN, TOK / BM), 256, 0, stream>>>(
            mb, fc1w_b, fc1b, nullptr, (void*)hb, TOK, HIDDEN, DIMC);

    // 8) out = x2 + hb @ fc2_w + fc2_b     (8192 x 1024, K=4096) -> f32
    gemm_bf16_wmma<false, true, true>
        <<<dim3(DIMC / BN, TOK / BM), 256, 0, stream>>>(
            hb, fc2w_b, fc2b, x2, (void*)out, TOK, DIMC, HIDDEN);
}
